// Low_Resnet_Generator_17179869783
// MI455X (gfx1250) — compile-verified, matrix operand order checked by reading, not run on hardware
//
#include <hip/hip_runtime.h>
#include <hip/hip_bf16.h>
#include <math.h>

typedef __attribute__((ext_vector_type(16))) __bf16 v16bf;
typedef __attribute__((ext_vector_type(8)))  float  v8f;
typedef __attribute__((ext_vector_type(4)))  unsigned int u32x4;
typedef __attribute__((ext_vector_type(8)))  int   i32x8;
typedef __attribute__((ext_vector_type(4)))  int   i32x4;

#define HW    65536          // 256*256
#define BATCH 16
#define NPIX  (BATCH * HW)   // 1048576 flattened pixels (N*H*W)
#define EPS   1e-5f

#if defined(__has_builtin)
#if __has_builtin(__builtin_amdgcn_tensor_load_to_lds) && __has_builtin(__builtin_amdgcn_s_wait_tensorcnt)
#define HAVE_TDM 1
#endif
#endif

__device__ __forceinline__ float celu1(float v) { return v > 0.f ? v : expm1f(v); }

// ISA 7.12.2 16-bit A/B fragment K-index for element e (0..15) of a lane:
// lanes 0-15 cover K {0..7,16..23}, lanes 16-31 cover K {8..15,24..31}
__device__ __forceinline__ int kmap(int e, int lane) {
  return (e & 7) + ((e >> 3) << 4) + ((lane >> 4) << 3);
}

// ---------------------------------------------------------------------------
// Tensor Data Mover: 2D tile load Global->LDS (D# per CDNA5 ISA ch.8).
// tile_w x tile_h elements of 4 bytes, row stride row_stride elements.
// ---------------------------------------------------------------------------
__device__ __forceinline__ void tdm_load_2d_f32(unsigned lds_addr, const void* gaddr,
                                                unsigned tile_w, unsigned tile_h,
                                                unsigned long long row_stride) {
#if HAVE_TDM
  const unsigned long long ga = (unsigned long long)(size_t)gaddr;
  u32x4 g0;
  g0[0] = 1u;                                           // count=1, user mode
  g0[1] = lds_addr;                                     // lds_addr [63:32]
  g0[2] = (unsigned)(ga & 0xFFFFFFFFu);                 // global_addr [95:64]
  g0[3] = (unsigned)((ga >> 32) & 0x1FFFFFFu) | (2u << 30);  // addr hi + type=2 (image)
  i32x8 g1;
  g1[0] = (int)(2u << 16);                              // wg_mask=0, data_size=4B
  g1[1] = (int)((tile_w & 0xFFFFu) << 16);              // tensor_dim0[15:0] @ [63:48]
  g1[2] = (int)((tile_w >> 16) | ((tile_h & 0xFFFFu) << 16)); // dim0 hi / dim1 lo
  g1[3] = (int)((tile_h >> 16) | ((tile_w & 0xFFFFu) << 16)); // dim1 hi / tile_dim0
  g1[4] = (int)(tile_h & 0xFFFFu);                      // tile_dim1 (tile_dim2=0)
  g1[5] = (int)(row_stride & 0xFFFFFFFFull);            // tensor_dim0_stride lo32
  g1[6] = (int)((row_stride >> 32) & 0xFFFFull);        // stride hi16, dim1_stride lo=0
  g1[7] = 0;
  i32x4 g2 = {0, 0, 0, 0}, g3 = {0, 0, 0, 0};
#if defined(__clang_major__) && (__clang_major__ >= 23)
  i32x8 g4 = {0, 0, 0, 0, 0, 0, 0, 0};
  __builtin_amdgcn_tensor_load_to_lds(g0, g1, g2, g3, g4, 0);
#else
  __builtin_amdgcn_tensor_load_to_lds(g0, g1, g2, g3, 0);
#endif
#endif
}

// ---------------------------------------------------------------------------
// Per-(n,c) mean / 1/std (ddof=1) for the instance-norm branch. 48 blocks.
// ---------------------------------------------------------------------------
__global__ void stats_nc_kernel(const float* __restrict__ x, float* __restrict__ mstats) {
  __shared__ float s1[256], s2[256];
  const int g = blockIdx.x;                 // n*3+c
  const float* p = x + (size_t)g * HW;
  float a = 0.f, b = 0.f;
  for (int i = threadIdx.x; i < HW; i += 256) { float v = p[i]; a += v; b += v * v; }
  s1[threadIdx.x] = a; s2[threadIdx.x] = b; __syncthreads();
  for (int st = 128; st > 0; st >>= 1) {
    if (threadIdx.x < st) { s1[threadIdx.x] += s1[threadIdx.x + st]; s2[threadIdx.x] += s2[threadIdx.x + st]; }
    __syncthreads();
  }
  if (threadIdx.x == 0) {
    float s = s1[0], ss = s2[0];
    float mean = s / (float)HW;
    float var  = (ss - s * mean) / (float)(HW - 1);   // ddof=1
    mstats[2 * g]     = mean;
    mstats[2 * g + 1] = rsqrtf(var);
  }
}

// ---------------------------------------------------------------------------
// 3x3 conv 6->64 (pad 1) implicit GEMM, K = 54 padded to 64. Stem only reads
// 12 MB so the register-gather im2col is fine here.
// ---------------------------------------------------------------------------
__global__ __launch_bounds__(256)
void conv3x3_wmma_kernel(const float* __restrict__ x, const float* __restrict__ mstats,
                         const float* __restrict__ w, const float* __restrict__ bias,
                         float* __restrict__ out) {
  const int lane = threadIdx.x & 31;
  const int wave = blockIdx.x * (blockDim.x >> 5) + (threadIdx.x >> 5);
  const int mt   = wave & 3;
  const int nt   = wave >> 2;
  const int m    = mt * 16 + (lane & 15);
  const int col  = nt * 16 + (lane & 15);
  const int n    = col >> 16;
  const int hw   = col & (HW - 1);
  const int y0   = hw >> 8;
  const int x0   = hw & 255;

  v8f acc = {};
  for (int kc = 0; kc < 64; kc += 32) {
    v16bf a, b;
#pragma unroll
    for (int e = 0; e < 16; ++e) {
      const int kk = kc + kmap(e, lane);
      float av = 0.f, bv = 0.f;
      if (kk < 54) {
        av = w[m * 54 + kk];
        const int c  = kk / 9;
        const int rs = kk - c * 9;
        const int r  = rs / 3;
        const int s  = rs - r * 3;
        const int yy = y0 + r - 1, xx = x0 + s - 1;
        if ((unsigned)yy < 256u && (unsigned)xx < 256u) {
          const int cc  = (c < 3) ? c : (c - 3);
          const float raw = x[(size_t)((n * 3 + cc) << 16) + (yy << 8) + xx];
          if (c < 3) bv = raw;
          else {
            const int g = n * 3 + cc;
            bv = (raw - mstats[2 * g]) * mstats[2 * g + 1];
          }
        }
      }
      a[e] = (__bf16)av;
      b[e] = (__bf16)bv;
    }
    acc = __builtin_amdgcn_wmma_f32_16x16x32_bf16(false, a, false, b, (short)0, acc, false, false);
  }
#pragma unroll
  for (int r = 0; r < 8; ++r) {
    const int mo = mt * 16 + r + ((lane >> 4) << 3);
    out[(size_t)(n * 64 + mo) * HW + hw] = acc[r] + bias[mo];
  }
}

// ---------------------------------------------------------------------------
// 1x1 conv as GEMM with LDS staging. Block = 256 threads = 8 waves owns a
// 256-pixel column tile. act(in) tile (CINP x 256) staged to LDS as bf16
// (TDM DMA when available, else coalesced float4 loads); weights staged as
// padded bf16. Waves iterate 16x16 WMMA tiles from LDS.
// ---------------------------------------------------------------------------
template <int CIN, int COUT, bool ACT>
__global__ __launch_bounds__(256)
void conv1x1_wmma_lds_kernel(const float* __restrict__ in, const float* __restrict__ coef,
                             const float* __restrict__ w, const float* __restrict__ bias,
                             float* __restrict__ out) {
  constexpr int MT   = COUT / 16;
  constexpr int CINP = (CIN < 32) ? 32 : CIN;       // K padded to WMMA depth
  __shared__ __bf16 Bs[CINP][256 + 8];              // activation tile
  __shared__ __bf16 Ws[COUT][CINP + 2];             // weight tile (pad kills bank conflicts)

  const int tid  = threadIdx.x;
  const int pix0 = blockIdx.x * 256;                // 256-aligned -> single n
  const int n    = pix0 >> 16;
  const int hw0  = pix0 & (HW - 1);
  const float* gsrc = in + (size_t)n * CIN * HW + hw0;

  // weights -> LDS (bf16, zero-padded K)
  for (int idx = tid; idx < COUT * CINP; idx += 256) {
    const int m = idx / CINP, c = idx % CINP;
    Ws[m][c] = (__bf16)((c < CIN) ? w[m * CIN + c] : 0.f);
  }

#if HAVE_TDM
  __shared__ float Stage[CIN][256];                 // raw fp32 tile via TDM
  if (tid == 0) {
    tdm_load_2d_f32((unsigned)(size_t)&Stage[0][0], gsrc, 256u, (unsigned)CIN,
                    (unsigned long long)HW);
    __builtin_amdgcn_s_wait_tensorcnt((short)0);
  }
  __syncthreads();
  for (int idx = tid; idx < CINP * 256; idx += 256) {
    const int c = idx >> 8, p = idx & 255;
    float v = (c < CIN) ? Stage[c][p] : 0.f;
    if (ACT && c < CIN) v = celu1(coef[c] * v + coef[CIN + c]);
    Bs[c][p] = (__bf16)v;
  }
  __syncthreads();
#else
  // coalesced fallback: 64 threads per channel row, float4 per thread
  for (int c = tid >> 6; c < CINP; c += 4) {
    const int p = (tid & 63) * 4;
    float4 v = make_float4(0.f, 0.f, 0.f, 0.f);
    if (c < CIN) v = *(const float4*)(gsrc + (size_t)c * HW + p);
    if (ACT && c < CIN) {
      const float sc = coef[c], sh = coef[CIN + c];
      v.x = celu1(sc * v.x + sh); v.y = celu1(sc * v.y + sh);
      v.z = celu1(sc * v.z + sh); v.w = celu1(sc * v.w + sh);
    }
    Bs[c][p + 0] = (__bf16)v.x; Bs[c][p + 1] = (__bf16)v.y;
    Bs[c][p + 2] = (__bf16)v.z; Bs[c][p + 3] = (__bf16)v.w;
  }
  __syncthreads();
#endif

  const int wv = tid >> 5, lane = tid & 31;
  for (int t = wv; t < MT * 16; t += 8) {
    const int mt   = t % MT;
    const int nt   = t / MT;
    const int colp = nt * 16 + (lane & 15);
    v8f acc = {};
#pragma unroll
    for (int kc = 0; kc < CINP; kc += 32) {
      v16bf a, b;
#pragma unroll
      for (int e = 0; e < 16; ++e) {
        const int c = kc + kmap(e, lane);
        a[e] = Ws[mt * 16 + (lane & 15)][c];
        b[e] = Bs[c][colp];
      }
      acc = __builtin_amdgcn_wmma_f32_16x16x32_bf16(false, a, false, b, (short)0, acc, false, false);
    }
    const int hw = hw0 + colp;
#pragma unroll
    for (int r = 0; r < 8; ++r) {
      const int mo = mt * 16 + r + ((lane >> 4) << 3);
      out[(size_t)(n * COUT + mo) * HW + hw] = acc[r] + bias[mo];
    }
  }
}

// ---------------------------------------------------------------------------
// Per-channel sum / sum-of-squares over (N,H,W).
// ---------------------------------------------------------------------------
__global__ void zero_kernel(float* p, int nfloats) {
  int i = blockIdx.x * blockDim.x + threadIdx.x;
  if (i < nfloats) p[i] = 0.f;
}

__global__ void chan_stats_kernel(const float* __restrict__ src, int C, float* __restrict__ sums) {
  __shared__ float s1[256], s2[256];
  const int ch = blockIdx.y;
  float a = 0.f, b = 0.f;
  for (int i = blockIdx.x * 256 + threadIdx.x; i < NPIX; i += gridDim.x * 256) {
    const int n = i >> 16, hw = i & (HW - 1);
    const float v = src[(size_t)(n * C + ch) * HW + hw];
    a += v; b += v * v;
  }
  s1[threadIdx.x] = a; s2[threadIdx.x] = b; __syncthreads();
  for (int st = 128; st > 0; st >>= 1) {
    if (threadIdx.x < st) { s1[threadIdx.x] += s1[threadIdx.x + st]; s2[threadIdx.x] += s2[threadIdx.x + st]; }
    __syncthreads();
  }
  if (threadIdx.x == 0) {
    atomicAdd(&sums[2 * ch],     s1[0]);
    atomicAdd(&sums[2 * ch + 1], s2[0]);
  }
}

__global__ void bn_coef_kernel(const float* __restrict__ sums, const float* __restrict__ gamma,
                               const float* __restrict__ beta, int C, float* __restrict__ coef) {
  const int t = threadIdx.x;
  if (t < C) {
    const float mean = sums[2 * t] / (float)NPIX;
    const float var  = sums[2 * t + 1] / (float)NPIX - mean * mean;  // biased (jnp.var)
    const float sc   = gamma[t] * rsqrtf(var + EPS);
    coef[t]     = sc;
    coef[C + t] = beta[t] - sc * mean;
  }
}

// ---------------------------------------------------------------------------
// Depthwise 3x3 (groups=16, pad 1) on act(in).
// ---------------------------------------------------------------------------
__global__ void dwconv3x3_kernel(const float* __restrict__ in, const float* __restrict__ coef,
                                 const float* __restrict__ w, const float* __restrict__ bias,
                                 float* __restrict__ out) {
  const int t    = blockIdx.x * 256 + threadIdx.x;   // 16*NPIX threads
  const int hw   = t & (HW - 1);
  const int rest = t >> 16;
  const int c    = rest & 15;
  const int n    = rest >> 4;
  const int y    = hw >> 8, x0 = hw & 255;
  const float sc = coef[c], sh = coef[16 + c];
  const float* base = in + (size_t)(n * 16 + c) * HW;
  float acc = bias[c];
#pragma unroll
  for (int r = 0; r < 3; ++r)
#pragma unroll
    for (int s = 0; s < 3; ++s) {
      const int yy = y + r - 1, xs = x0 + s - 1;
      if ((unsigned)yy < 256u && (unsigned)xs < 256u)
        acc += w[c * 9 + r * 3 + s] * celu1(sc * base[(yy << 8) + xs] + sh);
    }
  out[(size_t)(n * 16 + c) * HW + hw] = acc;
}

// ---------------------------------------------------------------------------
// Residual tail: out = celu(bn(pre)) + (ACTSKIP ? celu(bn(skip)) : skip).
// May run in-place (out == pre).
// ---------------------------------------------------------------------------
template <bool ACTSKIP>
__global__ void resadd_kernel(const float* pre, const float* __restrict__ coefPre,
                              const float* skip, const float* __restrict__ coefSkip,
                              float* out) {
  const int i = blockIdx.x * 256 + threadIdx.x;   // 64*NPIX
  const int c = (i >> 16) & 63;
  const float h = celu1(coefPre[c] * pre[i] + coefPre[64 + c]);
  float s = skip[i];
  if (ACTSKIP) s = celu1(coefSkip[c] * s + coefSkip[64 + c]);
  out[i] = h + s;
}

// ---------------------------------------------------------------------------
// Head: out = celu(1x1 conv 64->3 (yf) + b) + x.
// ---------------------------------------------------------------------------
__global__ void head_kernel(const float* __restrict__ yf, const float* __restrict__ w,
                            const float* __restrict__ b, const float* __restrict__ x,
                            float* __restrict__ out) {
  const int t  = blockIdx.x * 256 + threadIdx.x;  // NPIX
  const int n  = t >> 16, hw = t & (HW - 1);
  const float* base = yf + (size_t)n * 64 * HW + hw;
  __builtin_prefetch(base, 0, 0);
  float a0 = 0.f, a1 = 0.f, a2 = 0.f;
#pragma unroll 4
  for (int k = 0; k < 64; ++k) {
    const float v = base[(size_t)k * HW];
    a0 += w[k] * v; a1 += w[64 + k] * v; a2 += w[128 + k] * v;
  }
  const size_t o = (size_t)n * 3 * HW + hw;
  out[o]          = celu1(a0 + b[0]) + x[o];
  out[o + HW]     = celu1(a1 + b[1]) + x[o + HW];
  out[o + 2 * HW] = celu1(a2 + b[2]) + x[o + 2 * HW];
}

// ---------------------------------------------------------------------------
extern "C" void kernel_launch(void* const* d_in, const int* in_sizes, int n_in,
                              void* d_out, int out_size, void* d_ws, size_t ws_size,
                              hipStream_t stream) {
  auto F = [&](int i) { return (const float*)d_in[i]; };
  const float* x = F(0);
  // 1:c1_w 2:c1_b 3:bn1_g 4:bn1_b | blk1: 5..16 | blk2: 17..28 | 29:c2_w 30:c2_b

  float* ws        = (float*)d_ws;
  float* mstats    = ws;           // 96
  float* sums      = ws + 128;     // 128
  float* coef_c1   = ws + 256;     // {scale[64], shift[64]}
  float* coef_b1z1 = ws + 512;
  float* coef_b1z2 = ws + 640;
  float* coef_b1z3 = ws + 768;
  float* coef_b2z1 = ws + 1024;
  float* coef_b2z2 = ws + 1152;
  float* coef_b2z3 = ws + 1280;
  float* z1 = ws + 2048;                       // 16 * NPIX
  float* z2 = z1 + (size_t)16 * NPIX;          // 16 * NPIX
  float* y1 = z2 + (size_t)16 * NPIX;          // 64 * NPIX
  float* z3 = y1 + (size_t)64 * NPIX;          // 64 * NPIX

  auto bnstats = [&](const float* src, int C, const float* g, const float* be, float* coef) {
    zero_kernel<<<1, 256, 0, stream>>>(sums, 128);
    chan_stats_kernel<<<dim3(128, C), 256, 0, stream>>>(src, C, sums);
    bn_coef_kernel<<<1, 64, 0, stream>>>(sums, g, be, C, coef);
  };

  // stem
  stats_nc_kernel<<<48, 256, 0, stream>>>(x, mstats);
  conv3x3_wmma_kernel<<<32768, 256, 0, stream>>>(x, mstats, F(1), F(2), y1);
  bnstats(y1, 64, F(3), F(4), coef_c1);

  // residual block 1 (input = celu(bn(y1)) applied in the staging pass)
  conv1x1_wmma_lds_kernel<64, 16, true><<<4096, 256, 0, stream>>>(y1, coef_c1, F(5), F(6), z1);
  bnstats(z1, 16, F(7), F(8), coef_b1z1);
  dwconv3x3_kernel<<<(16 * NPIX) / 256, 256, 0, stream>>>(z1, coef_b1z1, F(9), F(10), z2);
  bnstats(z2, 16, F(11), F(12), coef_b1z2);
  conv1x1_wmma_lds_kernel<16, 64, true><<<4096, 256, 0, stream>>>(z2, coef_b1z2, F(13), F(14), z3);
  bnstats(z3, 64, F(15), F(16), coef_b1z3);
  resadd_kernel<true><<<(64 * NPIX) / 256, 256, 0, stream>>>(z3, coef_b1z3, y1, coef_c1, z3);
  // z3 now holds block-1 output y2 (activation domain)

  // residual block 2 (input already in activation domain -> ACT=false)
  conv1x1_wmma_lds_kernel<64, 16, false><<<4096, 256, 0, stream>>>(z3, coef_c1, F(17), F(18), z1);
  bnstats(z1, 16, F(19), F(20), coef_b2z1);
  dwconv3x3_kernel<<<(16 * NPIX) / 256, 256, 0, stream>>>(z1, coef_b2z1, F(21), F(22), z2);
  bnstats(z2, 16, F(23), F(24), coef_b2z2);
  conv1x1_wmma_lds_kernel<16, 64, true><<<4096, 256, 0, stream>>>(z2, coef_b2z2, F(25), F(26), y1);
  bnstats(y1, 64, F(27), F(28), coef_b2z3);
  resadd_kernel<false><<<(64 * NPIX) / 256, 256, 0, stream>>>(y1, coef_b2z3, z3, coef_c1, y1);
  // y1 now holds final trunk activation

  head_kernel<<<NPIX / 256, 256, 0, stream>>>(y1, F(29), F(30), x, (float*)d_out);
}
